// PAConv_32263794328119
// MI455X (gfx1250) — matrix operand order repaired and checked by reading, not compiled, not run on hardware
//
#include <hip/hip_runtime.h>

typedef _Float16 h16 __attribute__((ext_vector_type(16)));
typedef _Float16 h8  __attribute__((ext_vector_type(8)));
typedef float    f8  __attribute__((ext_vector_type(8)));
typedef float    f4  __attribute__((ext_vector_type(4)));
typedef unsigned int u32x4 __attribute__((ext_vector_type(4)));
typedef int          i32x8 __attribute__((ext_vector_type(8)));
typedef int          i32x4 __attribute__((ext_vector_type(4)));

#define NB    4
#define NC    64
#define NN    4096
#define NKN   16            // neighbors per point
#define NM    8
#define NOUT  64
#define NHID  16
#define NKTOT 65536         // N*K
#define TROWS 128           // rows per workgroup tile
#define AP    136           // padded LDS row stride for A (halves): 272B -> 4-bank skew
#define BP    136           // padded LDS row stride for B slice (halves)

#if defined(__gfx1250__) && __has_builtin(__builtin_amdgcn_tensor_load_to_lds)
#define USE_TDM 1
#else
#define USE_TDM 0
#endif

// ---------------- kernel 1: W (128x512 f32) -> Wt (512x128 f16), K-contiguous per column
__global__ void paconv_wprep(const float* __restrict__ W, _Float16* __restrict__ Wt) {
    int idx = blockIdx.x * blockDim.x + threadIdx.x;     // 16384 threads
    for (int i = 0; i < 4; ++i) {
        int e = idx * 4 + i;                             // 0..65535
        int c = e >> 9;                                  // 0..127 (K)
        int d = e & 511;                                 // 0..511 (col = m*64+out)
        Wt[d * 128 + c] = (_Float16)W[c * 512 + d];
    }
}

// ---------------- kernel 2: fused ScoreNet + WMMA GEMM + assign_score + BN/ReLU
__global__ __launch_bounds__(256)
void paconv_main(const float* __restrict__ inf, const float* __restrict__ gxyz,
                 const _Float16* __restrict__ Wt,
                 const float* __restrict__ s1w, const float* __restrict__ s1s,
                 const float* __restrict__ s1b, const float* __restrict__ s2w,
                 const float* __restrict__ s2b, const float* __restrict__ bns,
                 const float* __restrict__ bnb, float* __restrict__ out)
{
    // Af [128][AP] lives only until A fragments are in VGPRs; afterwards the same
    // region is reused for the per-m B slice (Bl, 17408B) and the output transpose
    // buffer (OutT, 32KB). Sc stays live through the m-loop.
    __shared__ alignas(16) unsigned char smem[TROWS * AP * 2 + TROWS * NM * 4]; // 38912 B
    _Float16* Af   = (_Float16*)smem;                    // [128][AP]: [0..63]=diff, [64..127]=identity
    float*    Sc   = (float*)(smem + TROWS * AP * 2);    // [128][8] softmax scores
    _Float16* Bl   = (_Float16*)smem;                    // [64][BP] per-m B slice (reuses Af)
    float*    OutT = (float*)smem;                       // [128][64] f32 (reuses Af)

    const int t = threadIdx.x;
    const int w = t >> 5;
    const int l = t & 31;
    const int rowg = blockIdx.x * TROWS;                 // global flattened (b,n,k) row
    const int b    = rowg >> 16;                         // / (N*K); tiles never straddle b
    const int rl   = rowg & (NKTOT - 1);                 // row within batch

    // ---- phase 1: stage raw in_feat tile f32->f16 into identity half (coalesced)
    for (int j = 0; j < 8; ++j) {
        int c = j * 8 + w;                               // each wave: 8 channels
        f4 v = *(const f4*)(inf + ((size_t)(b * NC + c) << 16) + rl + l * 4);
        int r0 = l * 4;
        Af[(r0 + 0) * AP + 64 + c] = (_Float16)v[0];
        Af[(r0 + 1) * AP + 64 + c] = (_Float16)v[1];
        Af[(r0 + 2) * AP + 64 + c] = (_Float16)v[2];
        Af[(r0 + 3) * AP + 64 + c] = (_Float16)v[3];
    }

    // ---- phase 1b: ScoreNet (3 -> 16 -> 8 -> softmax) for this tile's 128 points
    if (t < TROWS) {
        int rp = rl + t;
        int rc = rl + (t & ~(NKN - 1));                  // k=0 neighbor of the same point
        const float* gx = gxyz + ((size_t)(b * 3) << 16);
        float dx = gx[rp]             - gx[rc];
        float dy = gx[(1 << 16) + rp] - gx[(1 << 16) + rc];
        float dz = gx[(2 << 16) + rp] - gx[(2 << 16) + rc];
        float s[NM];
        for (int m = 0; m < NM; ++m) s[m] = s2b[m];
        for (int hh = 0; hh < NHID; ++hh) {
            float h = s1w[hh * 3] * dx + s1w[hh * 3 + 1] * dy + s1w[hh * 3 + 2] * dz;
            h = fmaxf(h * s1s[hh] + s1b[hh], 0.f);
            for (int m = 0; m < NM; ++m) s[m] += s2w[m * NHID + hh] * h;
        }
        float mx = s[0];
        for (int m = 1; m < NM; ++m) mx = fmaxf(mx, s[m]);
        float sum = 0.f;
        for (int m = 0; m < NM; ++m) { s[m] = __expf(s[m] - mx); sum += s[m]; }
        float inv = 1.f / sum;
        for (int m = 0; m < NM; ++m) Sc[t * NM + m] = s[m] * inv;
    }
    __syncthreads();

    // ---- phase 2: diff half Af[r][c] = Af[r][64+c] - Af[k0(r)][64+c] (packed f16)
    {
        int r  = t >> 1;
        int ch = (t & 1) * 32;
        int r0 = r & ~(NKN - 1);
        for (int i = 0; i < 4; ++i) {
            h8 a  = *(const h8*)(Af + r  * AP + 64 + ch + i * 8);
            h8 c0 = *(const h8*)(Af + r0 * AP + 64 + ch + i * 8);
            h8 d  = a - c0;
            *(h8*)(Af + r * AP + ch + i * 8) = d;
        }
    }
    __syncthreads();

    // ---- phase 3: hoist A fragments (16 rows x 128 ch) into VGPRs; Af is dead after this
    const bool hi   = (l >= 16);
    const int  arow = w * 16 + (l & 15);
    h16 A[4];
    for (int kb = 0; kb < 4; ++kb) {
        const _Float16* p = Af + arow * AP + kb * 32 + (hi ? 8 : 0);
        h8 a0 = *(const h8*)p;                           // K = kb*32 + {0..7 | 8..15}
        h8 a1 = *(const h8*)(p + 16);                    // K = kb*32 + {16..23 | 24..31}
        h16 av;
        for (int i = 0; i < 8; ++i) { av[i] = a0[i]; av[i + 8] = a1[i]; }
        A[kb] = av;
    }

    // ---- phase 4: m-serialized GEMM; B slice staged block-wide into LDS each m
    f8 fin[4] = {};                                      // 64 OUT columns, f32
    const int colb = (l & 15);
    for (int m = 0; m < NM; ++m) {
        __syncthreads();                                 // all waves done reading previous slice / Af
#if USE_TDM
        if (w == 0) {
            // 1-row TDM copy: 8192 f16 from Wt slice -> LDS, padding 16B every 256B
            // so LDS lands in the [64][BP=136] bank-skewed layout.
            unsigned long long ga = (unsigned long long)(const void*)(Wt + (size_t)m * 64 * 128);
            unsigned int lds = (unsigned int)(unsigned long long)(void*)Bl;
            u32x4 g0;
            g0[0] = 1u;                                          // count=1 (valid user D#)
            g0[1] = lds;                                         // lds_addr
            g0[2] = (unsigned int)(ga & 0xffffffffu);            // global_addr[31:0]
            g0[3] = (unsigned int)((ga >> 32) & 0x01ffffffu)     // global_addr[56:32]
                  | (2u << 30);                                  // type=2 ("image")
            i32x8 g1;
            g1[0] = (1 << 16)                                    // data_size = 2B
                  | (1 << 20)                                    // pad_enable
                  | (5 << 22)                                    // pad_interval: 2^5*8B = 256B
                  | (3 << 25);                                   // pad_amount: 4 DWORDs = 16B
            g1[1] = (int)((8192u & 0xffffu) << 16);              // tensor_dim0 lo16 (8192)
            g1[2] = (int)((8192u >> 16) | (1u << 16));           // tensor_dim0 hi16, tensor_dim1=1 lo16
            g1[3] = (int)(8192u << 16);                          // tensor_dim1 hi16=0, tile_dim0=8192
            g1[4] = 1;                                           // tile_dim1=1, tile_dim2=0
            g1[5] = 8192;                                        // tensor_dim0_stride lo32
            g1[6] = 0;                                           // stride hi16, dim1_stride lo16
            g1[7] = 0;
            i32x4 z4 = {0, 0, 0, 0};
            i32x8 z8 = {0, 0, 0, 0, 0, 0, 0, 0};
            __builtin_amdgcn_tensor_load_to_lds(g0, g1, z4, z4, z8, 0);
            __builtin_amdgcn_s_wait_tensorcnt(0);
        }
#else
        // fallback: cooperative copy into the same padded layout
        for (int i = 0; i < 4; ++i) {
            int e   = t + i * 256;                       // 1024 chunks of 8 halves
            int col = e >> 4;                            // 16 chunks per column
            int ko  = (e & 15) * 8;
            *(h8*)(Bl + col * BP + ko) =
                *(const h8*)(Wt + (size_t)m * 8192 + col * 128 + ko);
        }
#endif
        __syncthreads();                                 // B slice m visible to all waves

        f8 acc[4] = {};
        for (int ct = 0; ct < 4; ++ct) {
            const _Float16* bp = Bl + (ct * 16 + colb) * BP + (hi ? 8 : 0);
            for (int kb = 0; kb < 4; ++kb) {
                h8 b0 = *(const h8*)(bp + kb * 32);      // conflict-free ds_load_b128 pairs
                h8 b1 = *(const h8*)(bp + kb * 32 + 16);
                h16 Bv;
                for (int i = 0; i < 8; ++i) { Bv[i] = b0[i]; Bv[i + 8] = b1[i]; }
                acc[ct] = __builtin_amdgcn_wmma_f32_16x16x32_f16(
                    false, A[kb], false, Bv, (short)0, acc[ct], false, false);
            }
        }
        float sv[8];                                     // softmax score per accumulator row
        for (int v = 0; v < 8; ++v)
            sv[v] = Sc[(w * 16 + v + (hi ? 8 : 0)) * NM + m];
        for (int ct = 0; ct < 4; ++ct)
            for (int v = 0; v < 8; ++v)
                fin[ct][v] = fmaf(sv[v], acc[ct][v], fin[ct][v]);
    }
    __syncthreads();                                     // Bl/Sc dead -> reuse smem for OutT

    // ---- phase 5: BN + ReLU, transpose-stage through LDS
    for (int ct = 0; ct < 4; ++ct) {
        int d = ct * 16 + (l & 15);
        float sc = bns[d], bi = bnb[d];
        for (int v = 0; v < 8; ++v) {
            float val = fmaxf(fin[ct][v] * sc + bi, 0.f);
            int r = w * 16 + v + (hi ? 8 : 0);
            OutT[r * 64 + d] = val;
        }
    }
    __syncthreads();

    // ---- phase 6: 16B coalesced stores to (B,OUT,N,K)
    for (int j = 0; j < 8; ++j) {
        int id = j * 32 + l;
        int d  = id >> 2;                                // 0..63
        int rs = (id & 3) * 4;                           // 0,4,8,12
        int r  = w * 16 + rs;
        f4 vv;
        vv[0] = OutT[(r + 0) * 64 + d];
        vv[1] = OutT[(r + 1) * 64 + d];
        vv[2] = OutT[(r + 2) * 64 + d];
        vv[3] = OutT[(r + 3) * 64 + d];
        *(f4*)(out + ((size_t)(b * NOUT + d) << 16) + rl + r) = vv;
    }
}

extern "C" void kernel_launch(void* const* d_in, const int* in_sizes, int n_in,
                              void* d_out, int out_size, void* d_ws, size_t ws_size,
                              hipStream_t stream) {
    const float* inf  = (const float*)d_in[0];
    const float* gxyz = (const float*)d_in[1];
    const float* W    = (const float*)d_in[2];
    const float* s1w  = (const float*)d_in[3];
    const float* s1s  = (const float*)d_in[4];
    const float* s1b  = (const float*)d_in[5];
    const float* s2w  = (const float*)d_in[6];
    const float* s2b  = (const float*)d_in[7];
    const float* bns  = (const float*)d_in[8];
    const float* bnb  = (const float*)d_in[9];
    float*    out = (float*)d_out;
    _Float16* Wt  = (_Float16*)d_ws;                     // 512*128*2 = 128 KB

    paconv_wprep<<<64, 256, 0, stream>>>(W, Wt);
    paconv_main<<<(NB * NKTOT) / TROWS, 256, 0, stream>>>(
        inf, gxyz, Wt, s1w, s1s, s1b, s2w, s2b, bns, bnb, out);
}
